// EaRL_5188320493898
// MI455X (gfx1250) — matrix operand order repaired and checked by reading, not compiled, not run on hardware
//
#include <hip/hip_runtime.h>

// ---------------------------------------------------------------------------
// CDNA5 / gfx1250 hetero-GNN forward (SAGE x2 -> TransformerConv -> heads).
// Dense linears: v_wmma_f32_16x16x32_bf16 (bf16 mul, f32 accumulate), LDS
// staged as packed bf16 pairs, vectorized fast-path staging, global prefetch.
// Segment ops: f32 global atomics.
// ---------------------------------------------------------------------------

typedef __attribute__((ext_vector_type(16))) __bf16   v16bf;
typedef __attribute__((ext_vector_type(8)))  float    v8f;
typedef __attribute__((ext_vector_type(8)))  unsigned v8u;

#define TILE_K 32

// ----------------------------- utility kernels -----------------------------

__global__ void k_fill_f32(float* p, float v, long long n) {
  long long i = (long long)blockIdx.x * blockDim.x + threadIdx.x;
  if (i < n) p[i] = v;
}

__global__ void k_fill_u32(unsigned* p, unsigned v, long long n) {
  long long i = (long long)blockIdx.x * blockDim.x + threadIdx.x;
  if (i < n) p[i] = v;
}

__global__ void k_relu(float* x, long long n) {
  long long i = (long long)blockIdx.x * blockDim.x + threadIdx.x;
  if (i < n) x[i] = fmaxf(x[i], 0.0f);
}

__global__ void k_sigmoid(float* x, int n) {
  int i = blockIdx.x * blockDim.x + threadIdx.x;
  if (i < n) x[i] = 1.0f / (1.0f + __expf(-x[i]));
}

// out[i] = b[i%64] - W[i%64]   (dst features are all -1, input dim 1)
__global__ void k_rowdiff(const float* __restrict__ W, const float* __restrict__ b,
                          float* __restrict__ out, long long total) {
  long long i = (long long)blockIdx.x * blockDim.x + threadIdx.x;
  if (i < total) {
    int j = (int)(i & 63);
    out[i] = b[j] - W[j];
  }
}

// ------------------------- segment mean (SAGE aggr) ------------------------

__global__ void k_seg_accum(const float* __restrict__ x,
                            const int* __restrict__ src,
                            const int* __restrict__ dst,
                            float* __restrict__ sums, float* __restrict__ cnt,
                            int E, int F) {
  long long i = (long long)blockIdx.x * blockDim.x + threadIdx.x;
  long long total = (long long)E * F;
  if (i >= total) return;
  int e = (int)(i / F), f = (int)(i % F);
  int s = src[e], d = dst[e];
  atomicAdd(&sums[(long long)d * F + f], x[(long long)s * F + f]);
  if (f == 0) atomicAdd(&cnt[d], 1.0f);
}

__global__ void k_div_cnt(float* __restrict__ sums, const float* __restrict__ cnt,
                          int Nd, int F) {
  long long i = (long long)blockIdx.x * blockDim.x + threadIdx.x;
  long long total = (long long)Nd * F;
  if (i >= total) return;
  int d = (int)(i / F);
  sums[i] /= fmaxf(cnt[d], 1.0f);
}

// ------------------------------ WMMA GEMM ----------------------------------
// C[M,N] (+)= A[M,K] @ W[K,N] (+ bias).  Block = 128 threads (4 waves).
// Block tile: 32 rows x 64 cols; wave w owns cols [w*16, w*16+16) and issues
// TWO v_wmma_f32_16x16x32_bf16 per k-step (rows 0-15, 16-31) sharing one B
// fragment.  LDS holds bf16 PAIRS as 32-bit words:
//   As[row][wk]  : row-major,   wk = k/2   (32 rows x 16 words)
//   Bs[col][wk]  : col-major,   wk = k/2   (64 cols x 16 words)
// Fragment packing follows ISA 7.12.2 (16-bit A 16x32, 16-bit B 32x16).

__device__ inline unsigned pack_bf16(float a, float b) {
  union { unsigned u; __bf16 h[2]; } t;
  t.h[0] = (__bf16)a; t.h[1] = (__bf16)b;   // -> v_cvt_pk_bf16_f32
  return t.u;
}

__global__ __launch_bounds__(128)
void k_gemm_wmma(const float* __restrict__ A, const float* __restrict__ W,
                 const float* __restrict__ bias, float* __restrict__ C,
                 int M, int K, int N, int addC) {
  __shared__ unsigned As[32 * 16];   // 32 rows x 16 k-pair words
  __shared__ unsigned Bs[64 * 16];   // 64 cols x 16 k-pair words

  const int tid  = threadIdx.x;
  const int wave = tid >> 5;
  const int lane = tid & 31;
  const int m0   = blockIdx.x * 32;
  const int nb   = blockIdx.y * 64;
  const int half = lane >> 4;
  const int r16  = lane & 15;

  v8f acc0 = {}, acc1 = {};

  for (int k0 = 0; k0 < K; k0 += TILE_K) {
    // ---- stage A tile (32x32 f32 -> 512 packed words, 4 per thread) ----
    if (m0 + 32 <= M && k0 + TILE_K <= K) {
      for (int i = tid; i < 512; i += 128) {
        int r = i >> 4, wk = i & 15;
        const float2 f = *(const float2*)&A[(long long)(m0 + r) * K + k0 + wk * 2];
        As[i] = pack_bf16(f.x, f.y);
      }
    } else {
      for (int i = tid; i < 512; i += 128) {
        int r = i >> 4, wk = i & 15;
        int gr = m0 + r, gk = k0 + wk * 2;
        float a = (gr < M && gk     < K) ? A[(long long)gr * K + gk]     : 0.0f;
        float b = (gr < M && gk + 1 < K) ? A[(long long)gr * K + gk + 1] : 0.0f;
        As[i] = pack_bf16(a, b);
      }
    }
    // ---- stage W tile (32x64 f32 -> 1024 packed words, col-major) ----
    if (nb + 64 <= N && k0 + TILE_K <= K) {
      for (int i = tid; i < 1024; i += 128) {
        int c = i & 63, wk = i >> 6;                 // consecutive tid -> consecutive c
        float a = W[(long long)(k0 + 2 * wk)     * N + nb + c];
        float b = W[(long long)(k0 + 2 * wk + 1) * N + nb + c];
        Bs[c * 16 + wk] = pack_bf16(a, b);
      }
    } else {
      for (int i = tid; i < 1024; i += 128) {
        int c = i & 63, wk = i >> 6;
        int gn = nb + c, gk = k0 + 2 * wk;
        float a = (gn < N && gk     < K) ? W[(long long)gk * N + gn]       : 0.0f;
        float b = (gn < N && gk + 1 < K) ? W[(long long)(gk + 1) * N + gn] : 0.0f;
        Bs[c * 16 + wk] = pack_bf16(a, b);
      }
    }
    __syncthreads();

    // speculative prefetch of next A tile -> global_prefetch_b8
    if (k0 + TILE_K < K && m0 + (tid >> 2) < M) {
      __builtin_prefetch(&A[(long long)(m0 + (tid >> 2)) * K + k0 + TILE_K], 0, 1);
    }

    // ---- fragments: 8 ds_load_b32 each, bit-cast to v16bf ----
    union { v8u u; v16bf h; } fa0, fa1, fb;
#pragma unroll
    for (int p = 0; p < 8; ++p) {
      const int wa = (p < 4 ? 0 : 8) + half * 4 + (p & 3);  // A k-pair word
      fa0.u[p] = As[r16 * 16 + wa];
      fa1.u[p] = As[(r16 + 16) * 16 + wa];
      fb.u[p]  = Bs[(wave * 16 + r16) * 16 + wa];           // B k-pair word
    }
    acc0 = __builtin_amdgcn_wmma_f32_16x16x32_bf16(
        false, fa0.h, false, fb.h, (short)0, acc0, false, false);
    acc1 = __builtin_amdgcn_wmma_f32_16x16x32_bf16(
        false, fa1.h, false, fb.h, (short)0, acc1, false, false);
    __syncthreads();
  }

  // ---- epilogue: lane l, vgpr v -> row = half*8 + v, col = r16 ----
  const int col = nb + wave * 16 + r16;
  if (col < N) {
    float bv = bias ? bias[col] : 0.0f;
#pragma unroll
    for (int vv = 0; vv < 8; ++vv) {
      int rowg = m0 + half * 8 + vv;
      if (rowg < M) {
        long long idx = (long long)rowg * N + col;
        float o = acc0[vv] + bv;
        if (addC) o += C[idx];
        C[idx] = o;
      }
      int rowg1 = rowg + 16;
      if (rowg1 < M) {
        long long idx = (long long)rowg1 * N + col;
        float o = acc1[vv] + bv;
        if (addC) o += C[idx];
        C[idx] = o;
      }
    }
  }
}

// ---------------------- TransformerConv edge kernels -----------------------

__device__ inline unsigned fenc(float f) {
  unsigned u = __float_as_uint(f);
  return (u & 0x80000000u) ? ~u : (u | 0x80000000u);   // order-preserving map
}
__device__ inline float fdec(unsigned u) {
  unsigned v = (u & 0x80000000u) ? (u & 0x7fffffffu) : ~u;
  return __uint_as_float(v);
}
#define FENC_NEG_INF 0x007fffffu   // fenc(-inf)

__global__ void k_tc_logits(const float* __restrict__ q, int qstride,
                            const float* __restrict__ kk,
                            const int* __restrict__ src, const int* __restrict__ dst,
                            float* __restrict__ logits, unsigned* __restrict__ menc,
                            int E) {
  int e = blockIdx.x * blockDim.x + threadIdx.x;
  if (e >= E) return;
  int s = src[e], d = dst[e];
  const float* qp = q + (long long)d * qstride;
  const float* kp = kk + (long long)s * 64;
  const float inv = 0.1767766952966369f; // 1/sqrt(32)
#pragma unroll
  for (int h = 0; h < 2; ++h) {
    float a = 0.0f;
    for (int c = 0; c < 32; ++c) a += qp[h * 32 + c] * kp[h * 32 + c];
    float lg = a * inv;
    logits[(long long)e * 2 + h] = lg;
    atomicMax(&menc[(long long)d * 2 + h], fenc(lg));
  }
}

__global__ void k_tc_expsum(const float* __restrict__ logits,
                            const int* __restrict__ dst,
                            const unsigned* __restrict__ menc,
                            float* __restrict__ ex, float* __restrict__ den, int E) {
  int e = blockIdx.x * blockDim.x + threadIdx.x;
  if (e >= E) return;
  int d = dst[e];
#pragma unroll
  for (int h = 0; h < 2; ++h) {
    float m = fdec(menc[(long long)d * 2 + h]);
    if (!isfinite(m)) m = 0.0f;
    float v = __expf(logits[(long long)e * 2 + h] - m);
    ex[(long long)e * 2 + h] = v;
    atomicAdd(&den[(long long)d * 2 + h], v);
  }
}

__global__ void k_tc_scatter(const float* __restrict__ ex, const float* __restrict__ den,
                             const float* __restrict__ vv,
                             const int* __restrict__ src, const int* __restrict__ dst,
                             float* __restrict__ out, int E) {
  long long i = (long long)blockIdx.x * blockDim.x + threadIdx.x;
  long long total = (long long)E * 64;
  if (i >= total) return;
  int e = (int)(i >> 6), j = (int)(i & 63), h = j >> 5;
  int d = dst[e], s = src[e];
  float a = ex[(long long)e * 2 + h] / fmaxf(den[(long long)d * 2 + h], 1e-16f);
  atomicAdd(&out[(long long)d * 64 + j], a * vv[(long long)s * 64 + j]);
}

// ----------------------- Bernoulli sample (threefry) -----------------------

__device__ inline unsigned rotl32(unsigned x, int d) { return (x << d) | (x >> (32 - d)); }

__global__ void k_bernoulli(const float* __restrict__ p, float* __restrict__ out, int n) {
  int i = blockIdx.x * blockDim.x + threadIdx.x;
  if (i >= n) return;
  unsigned k0 = 0u, k1 = 42u, ks2 = 0x1BD11BDAu ^ k0 ^ k1;
  unsigned x0 = 0u, x1 = (unsigned)i;
  const int R0[4] = {13, 15, 26, 6}, R1[4] = {17, 29, 16, 24};
  x0 += k0; x1 += k1;
#pragma unroll
  for (int r = 0; r < 5; ++r) {
    const int* R = (r & 1) ? R1 : R0;
#pragma unroll
    for (int j = 0; j < 4; ++j) { x0 += x1; x1 = rotl32(x1, R[j]); x1 ^= x0; }
    switch (r) {
      case 0: x0 += k1;  x1 += ks2 + 1u; break;
      case 1: x0 += ks2; x1 += k0 + 2u;  break;
      case 2: x0 += k0;  x1 += k1 + 3u;  break;
      case 3: x0 += k1;  x1 += ks2 + 4u; break;
      case 4: x0 += ks2; x1 += k0 + 5u;  break;
    }
  }
  float u = __uint_as_float((x0 >> 9) | 0x3f800000u) - 1.0f;
  out[i] = (u < p[i]) ? 1.0f : 0.0f;
}

// ------------------------------- host side ---------------------------------

struct RelT { int s, d, e; };
static const int  kNT       = 5;
static const int  kN[kNT]   = {5000, 100000, 30000, 15000, 15000};
static const int  kFin[kNT] = {16, 16, 64, 64, 8};
static const RelT kRels[13] = {
  {0,1,150000},{1,0,150000},{0,2,40000},{2,0,40000},{1,2,300000},{2,1,300000},
  {3,3,200000},{3,2,100000},{2,3,100000},{3,2,60000},{2,3,60000},{3,4,15000},{4,3,15000}
};

static inline int nblk(long long n, int b) { return (int)((n + b - 1) / b); }

extern "C" void kernel_launch(void* const* d_in, const int* in_sizes, int n_in,
                              void* d_out, int out_size, void* d_ws, size_t ws_size,
                              hipStream_t stream) {
  (void)in_sizes; (void)n_in; (void)out_size; (void)ws_size;

  // --- flattened input layout (setup_inputs() dict insertion order) ---
  const int sage0 = 5;             // convs[0]: 13 * (Wl,bl,Wr)
  const int sage1 = sage0 + 39;    // convs[1]
  const int tr2   = sage1 + 39;    // convs[2]: 13 * (Wq,bq,Wk,bk,Wv,bv,Ws,bs)
  const int plyr  = tr2 + 104;     // protein_layer: 8
  const int zmlp  = plyr + 8;      // zero MLP: 3*(W,b)
  const int vmlp  = zmlp + 6;      // value MLP: 3*(W,b)
  const int edg0  = vmlp + 6;      // 13 edge arrays, each (2,E) flat

  auto F = [&](int i) { return (const float*)d_in[i]; };

  // --- workspace bump allocator (reset each call: deterministic) ---
  char*  ws  = (char*)d_ws;
  size_t off = 0;
  auto alloc_f = [&](long long n) -> float* {
    float* p = (float*)(ws + off);
    off += (size_t)(((n * 4 + 255) / 256) * 256);
    return p;
  };

  float* bankA[kNT]; float* bankB[kNT];
  for (int t = 0; t < kNT; ++t) bankA[t] = alloc_f((long long)kN[t] * 128);
  for (int t = 0; t < kNT; ++t) bankB[t] = alloc_f((long long)kN[t] * 128);
  float*    agg    = alloc_f((long long)100000 * 128);
  float*    cnt    = alloc_f(100000);
  float*    qbuf   = alloc_f((long long)100000 * 64);
  float*    kbuf   = alloc_f((long long)100000 * 64);
  float*    vbuf   = alloc_f((long long)100000 * 64);
  float*    logits = alloc_f((long long)300000 * 2);
  float*    exbuf  = alloc_f((long long)300000 * 2);
  unsigned* menc   = (unsigned*)alloc_f((long long)100000 * 2);
  float*    den    = alloc_f((long long)100000 * 2);
  float*    qrow   = alloc_f(64);
  float*    pnbuf  = alloc_f((long long)15000 * 64);
  float*    t1     = alloc_f((long long)15000 * 512);
  float*    t2     = alloc_f((long long)15000 * 512);

  auto fillf = [&](float* p, float v, long long n) {
    k_fill_f32<<<nblk(n, 256), 256, 0, stream>>>(p, v, n);
  };
  auto fillu = [&](unsigned* p, unsigned v, long long n) {
    k_fill_u32<<<nblk(n, 256), 256, 0, stream>>>(p, v, n);
  };
  auto gemm = [&](const float* A, const float* W, const float* bias, float* C,
                  int M, int K, int N, int addC) {
    dim3 grid((unsigned)((M + 31) / 32), (unsigned)((N + 63) / 64));
    k_gemm_wmma<<<grid, 128, 0, stream>>>(A, W, bias, C, M, K, N, addC);
  };

  const float* xcur[kNT];
  for (int t = 0; t < kNT; ++t) xcur[t] = F(t);
  float** nextbank  = bankA;
  float** otherbank = bankB;
  int dims[kNT] = {kFin[0], kFin[1], kFin[2], kFin[3], kFin[4]};

  // ---------------- layers 0,1: hetero SAGE -> 128, ReLU ----------------
  for (int li = 0; li < 2; ++li) {
    const int base = (li == 0) ? sage0 : sage1;
    for (int t = 0; t < kNT; ++t) fillf(nextbank[t], 0.0f, (long long)kN[t] * 128);
    for (int r = 0; r < 13; ++r) {
      const int s = kRels[r].s, d = kRels[r].d, E = kRels[r].e;
      const int* src = (const int*)d_in[edg0 + r];
      const int* dst = src + E;
      const int Fs = dims[s], Fd = dims[d];
      fillf(agg, 0.0f, (long long)kN[d] * Fs);
      fillf(cnt, 0.0f, kN[d]);
      k_seg_accum<<<nblk((long long)E * Fs, 256), 256, 0, stream>>>(
          xcur[s], src, dst, agg, cnt, E, Fs);
      k_div_cnt<<<nblk((long long)kN[d] * Fs, 256), 256, 0, stream>>>(agg, cnt, kN[d], Fs);
      gemm(agg,     F(base + r * 3 + 0), F(base + r * 3 + 1), nextbank[d], kN[d], Fs, 128, 1);
      gemm(xcur[d], F(base + r * 3 + 2), nullptr,             nextbank[d], kN[d], Fd, 128, 1);
    }
    for (int t = 0; t < kNT; ++t)
      k_relu<<<nblk((long long)kN[t] * 128, 256), 256, 0, stream>>>(
          nextbank[t], (long long)kN[t] * 128);
    for (int t = 0; t < kNT; ++t) { xcur[t] = nextbank[t]; dims[t] = 128; }
    float** tmp = nextbank; nextbank = otherbank; otherbank = tmp;
  }

  // ------------- layer 2: hetero TransformerConv -> 64, ReLU -------------
  for (int t = 0; t < kNT; ++t) fillf(nextbank[t], 0.0f, (long long)kN[t] * 64);
  for (int r = 0; r < 13; ++r) {
    const int s = kRels[r].s, d = kRels[r].d, E = kRels[r].e;
    const int* src = (const int*)d_in[edg0 + r];
    const int* dst = src + E;
    const int pb = tr2 + r * 8; // Wq,bq,Wk,bk,Wv,bv,Ws,bs
    gemm(xcur[d], F(pb + 0), F(pb + 1), qbuf, kN[d], 128, 64, 0);
    gemm(xcur[s], F(pb + 2), F(pb + 3), kbuf, kN[s], 128, 64, 0);
    gemm(xcur[s], F(pb + 4), F(pb + 5), vbuf, kN[s], 128, 64, 0);
    fillu(menc, FENC_NEG_INF, (long long)kN[d] * 2);
    fillf(den, 0.0f, (long long)kN[d] * 2);
    k_tc_logits<<<nblk(E, 256), 256, 0, stream>>>(qbuf, 64, kbuf, src, dst, logits, menc, E);
    k_tc_expsum<<<nblk(E, 256), 256, 0, stream>>>(logits, dst, menc, exbuf, den, E);
    gemm(xcur[d], F(pb + 6), F(pb + 7), nextbank[d], kN[d], 128, 64, 1); // root skip
    k_tc_scatter<<<nblk((long long)E * 64, 256), 256, 0, stream>>>(
        exbuf, den, vbuf, src, dst, nextbank[d], E);
  }
  for (int t = 0; t < kNT; ++t)
    k_relu<<<nblk((long long)kN[t] * 64, 256), 256, 0, stream>>>(
        nextbank[t], (long long)kN[t] * 64);
  for (int t = 0; t < kNT; ++t) xcur[t] = nextbank[t];

  // --------- protein_layer: TransformerConv with x_dst = -1 (dim 1) ---------
  {
    const int r = 11; // protein --is_named--> protein_name
    const int E = kRels[r].e;
    const int* src = (const int*)d_in[edg0 + r];
    const int* dst = src + E;
    k_rowdiff<<<1, 64, 0, stream>>>(F(plyr + 0), F(plyr + 1), qrow, 64); // q = bq - Wq
    gemm(xcur[3], F(plyr + 2), F(plyr + 3), kbuf, kN[3], 64, 64, 0);
    gemm(xcur[3], F(plyr + 4), F(plyr + 5), vbuf, kN[3], 64, 64, 0);
    fillu(menc, FENC_NEG_INF, (long long)kN[4] * 2);
    fillf(den, 0.0f, (long long)kN[4] * 2);
    k_tc_logits<<<nblk(E, 256), 256, 0, stream>>>(qrow, 0, kbuf, src, dst, logits, menc, E);
    k_tc_expsum<<<nblk(E, 256), 256, 0, stream>>>(logits, dst, menc, exbuf, den, E);
    k_rowdiff<<<nblk((long long)kN[4] * 64, 256), 256, 0, stream>>>(
        F(plyr + 6), F(plyr + 7), pnbuf, (long long)kN[4] * 64); // skip = bs - Ws
    k_tc_scatter<<<nblk((long long)E * 64, 256), 256, 0, stream>>>(
        exbuf, den, vbuf, src, dst, pnbuf, E);
  }

  // ------------------------------- heads -------------------------------
  float* outp   = (float*)d_out;
  float* p_zero = outp;           // [0, 15000)
  float* sample = outp + 15000;   // [15000, 30000)
  float* value  = outp + 30000;   // [30000, 45000)
  const int Np = kN[4];

  gemm(pnbuf, F(zmlp + 0), F(zmlp + 1), t1, Np, 64, 512, 0);
  k_relu<<<nblk((long long)Np * 512, 256), 256, 0, stream>>>(t1, (long long)Np * 512);
  gemm(t1, F(zmlp + 2), F(zmlp + 3), t2, Np, 512, 512, 0);
  k_relu<<<nblk((long long)Np * 512, 256), 256, 0, stream>>>(t2, (long long)Np * 512);
  gemm(t2, F(zmlp + 4), F(zmlp + 5), p_zero, Np, 512, 1, 0);
  k_sigmoid<<<nblk(Np, 256), 256, 0, stream>>>(p_zero, Np);

  k_bernoulli<<<nblk(Np, 256), 256, 0, stream>>>(p_zero, sample, Np);

  gemm(pnbuf, F(vmlp + 0), F(vmlp + 1), t1, Np, 64, 512, 0);
  k_relu<<<nblk((long long)Np * 512, 256), 256, 0, stream>>>(t1, (long long)Np * 512);
  gemm(t1, F(vmlp + 2), F(vmlp + 3), t2, Np, 512, 512, 0);
  k_relu<<<nblk((long long)Np * 512, 256), 256, 0, stream>>>(t2, (long long)Np * 512);
  gemm(t2, F(vmlp + 4), F(vmlp + 5), value, Np, 512, 1, 0);
}
